// SwinTransformer3D_52252572123374
// MI455X (gfx1250) — compile-verified, matrix operand order checked by reading, not run on hardware
//
#include <hip/hip_runtime.h>

// =====================================================================
// Swin-3D block for gfx1250 (MI455X). All GEMMs via v_wmma_f32_16x16x32_bf16.
// Weight tiles staged to LDS with global_load_async_to_lds_b128 (ASYNCcnt,
// ping-pong double buffer). Requires ws_size >= 256 MiB.
// =====================================================================

typedef __bf16        v16bf __attribute__((ext_vector_type(16)));
typedef float         v8f   __attribute__((ext_vector_type(8)));
typedef unsigned int  u32x4 __attribute__((ext_vector_type(4)));
typedef float         f32x4 __attribute__((ext_vector_type(4)));

union Frag { u32x4 q[2]; v16bf v; };   // 32 bytes = 16 bf16 = one A/B operand

__device__ __forceinline__ unsigned short f2bf(float f) {
    unsigned int u = __float_as_uint(f);
    u += 0x7FFFu + ((u >> 16) & 1u);           // round-to-nearest-even
    return (unsigned short)(u >> 16);
}

#define TOK   50176      // B*D*H*W = 2*8*56*56
#define CCH   256
#define NHEAD 8
#define HDIM  32
#define NWTOK 98         // tokens per window
#define NWINB 256        // windows per batch image

// ---------------------------------------------------------------------
// token(window-order) -> flat spatial index with cyclic (un)shift.
// Identical for forward shift (LN1 gather) and reverse scatter (proj).
// ---------------------------------------------------------------------
__device__ __forceinline__ size_t win_token_to_flat(int t) {
    int b_ = t / NWTOK, i = t % NWTOK;
    int b  = b_ >> 8;                 // /256
    int widx = b_ & 255;
    int wd = widx >> 6, wh = (widx >> 3) & 7, ww = widx & 7;
    int id = i / 49, ih = (i / 7) % 7, iw = i % 7;
    int d = (wd * 2 + id + 1) & 7;
    int h = (wh * 7 + ih + 3) % 56;
    int w = (ww * 7 + iw + 3) % 56;
    return (size_t)(((b * 8 + d) * 56 + h) * 56 + w);
}

// ---------------------------------------------------------------------
// fp32 -> bf16 weight conversion
// ---------------------------------------------------------------------
__global__ void f2bf_kernel(const float* __restrict__ src,
                            unsigned short* __restrict__ dst, int n) {
    int i = blockIdx.x * 256 + threadIdx.x;
    if (i < n) dst[i] = f2bf(src[i]);
}

// ---------------------------------------------------------------------
// LayerNorm (+ optional shift+window-partition gather) -> bf16 rows.
// One wave per token (C=256 -> 8 floats/lane), shuffle reductions.
// ---------------------------------------------------------------------
template<bool SHIFT>
__global__ __launch_bounds__(256) void ln_kernel(
    const float* __restrict__ x, const float* __restrict__ gamma,
    const float* __restrict__ beta, unsigned short* __restrict__ out)
{
    int lane  = threadIdx.x & 31;
    int token = blockIdx.x * 8 + (threadIdx.x >> 5);
    size_t src = SHIFT ? win_token_to_flat(token) : (size_t)token;

    const float* px = x + src * CCH + lane * 8;
    f32x4 a = *(const f32x4*)px;
    f32x4 b = *(const f32x4*)(px + 4);
    float vv[8] = {a.x, a.y, a.z, a.w, b.x, b.y, b.z, b.w};

    float s = 0.f, q = 0.f;
    #pragma unroll
    for (int k = 0; k < 8; ++k) { s += vv[k]; q += vv[k] * vv[k]; }
    #pragma unroll
    for (int m = 1; m < 32; m <<= 1) {
        s += __shfl_xor(s, m, 32);
        q += __shfl_xor(q, m, 32);
    }
    float mean = s * (1.f / 256.f);
    float var  = q * (1.f / 256.f) - mean * mean;
    float rstd = rsqrtf(var + 1e-5f);

    const float* pg = gamma + lane * 8;
    const float* pb = beta  + lane * 8;
    unsigned int pk[4];
    #pragma unroll
    for (int k = 0; k < 4; ++k) {
        float y0 = (vv[2*k]   - mean) * rstd * pg[2*k]   + pb[2*k];
        float y1 = (vv[2*k+1] - mean) * rstd * pg[2*k+1] + pb[2*k+1];
        pk[k] = (unsigned int)f2bf(y0) | ((unsigned int)f2bf(y1) << 16);
    }
    u32x4 o = {pk[0], pk[1], pk[2], pk[3]};
    *(u32x4*)(out + (size_t)token * CCH + lane * 8) = o;
}

// ---------------------------------------------------------------------
// Generic bf16 GEMM: C = A[M,K] @ Bw[N,K]^T + bias, WMMA-tiled.
// Block: 128 thr (4 waves). Block tile 128M x 64N; wave tile 32M x 64N.
// B tile (shared by all 4 waves) is staged to LDS each K-step with
// global_load_async_to_lds_b128, double-buffered so the copy of step
// k+1 overlaps the 8 WMMAs of step k (ASYNCcnt in-order completion:
// waiting to <=2 leaves exactly the newest buffer's copies in flight).
// MODE 0: bf16 out (QKV)         MODE 1: bf16 out + exact GELU (fc1)
// MODE 2: fp32 out, window-reverse+unshift scatter + residual (proj)
// MODE 3: fp32 out + residual in natural order (fc2 -> d_out)
// ---------------------------------------------------------------------
template<int MODE>
__global__ __launch_bounds__(128) void gemm_bf16(
    const unsigned short* __restrict__ A,   // M x K, bf16 row-major
    const unsigned short* __restrict__ Bw,  // N x K, bf16 row-major (over K)
    const float* __restrict__ bias,         // N
    int M, int N, int K,
    unsigned short* __restrict__ outb,
    float* __restrict__ outf,
    const float* __restrict__ resid)
{
    __shared__ __attribute__((aligned(16))) unsigned short btile[2][64 * 32];

    int tid  = threadIdx.x;
    int wave = tid >> 5;
    int lane = tid & 31;
    int sel  = lane >> 4, lm = lane & 15;
    int m0 = blockIdx.x * 128 + wave * 32;
    int n0 = blockIdx.y * 64;

    // Low 32 bits of a generic LDS pointer == wave-relative LDS offset.
    unsigned ldsbase = (unsigned)(unsigned long long)&btile[0][0];
    int chunk0 = tid * 2;                  // 256 x 16B chunks per 4KB tile

    auto stage = [&](int kk, int buf) {
        #pragma unroll
        for (int c = 0; c < 2; ++c) {
            int chunk = chunk0 + c;
            int row   = chunk >> 2;        // 0..63 : B column (n0+row)
            int part  = chunk & 3;         // 16B sub-chunk along K
            const unsigned short* src = Bw + (size_t)(n0 + row) * K + kk + part * 8;
            unsigned dst = ldsbase + (unsigned)(buf * 4096 + row * 64 + part * 16);
            asm volatile("global_load_async_to_lds_b128 %0, %1, off"
                         :: "v"(dst), "v"((unsigned long long)src)
                         : "memory");
        }
    };

    v8f acc[2][4] = {};
    const unsigned short* arow0 = A + (size_t)(m0 + lm) * K + sel * 8;
    const unsigned short* arow1 = A + (size_t)(m0 + 16 + lm) * K + sel * 8;

    stage(0, 0);
    int cur = 0;
    for (int kk = 0; kk < K; kk += 32) {
        if (kk + 32 < K) {
            stage(kk + 32, cur ^ 1);                       // prefetch next
            asm volatile("s_wait_asynccnt 0x2" ::: "memory");  // cur landed
        } else {
            asm volatile("s_wait_asynccnt 0x0" ::: "memory");
        }
        __syncthreads();

        Frag af0, af1;
        af0.q[0] = *(const u32x4*)(arow0 + kk);
        af0.q[1] = *(const u32x4*)(arow0 + kk + 16);
        af1.q[0] = *(const u32x4*)(arow1 + kk);
        af1.q[1] = *(const u32x4*)(arow1 + kk + 16);
        const unsigned short* bt = &btile[cur][0];
        #pragma unroll
        for (int nt = 0; nt < 4; ++nt) {
            Frag bf;
            const unsigned short* br = bt + (nt * 16 + lm) * 32 + sel * 8;
            bf.q[0] = *(const u32x4*)br;
            bf.q[1] = *(const u32x4*)(br + 16);
            acc[0][nt] = __builtin_amdgcn_wmma_f32_16x16x32_bf16(
                false, af0.v, false, bf.v, (short)0, acc[0][nt], false, false);
            acc[1][nt] = __builtin_amdgcn_wmma_f32_16x16x32_bf16(
                false, af1.v, false, bf.v, (short)0, acc[1][nt], false, false);
        }
        __syncthreads();
        cur ^= 1;
    }

    #pragma unroll
    for (int mt = 0; mt < 2; ++mt) {
        #pragma unroll
        for (int r = 0; r < 8; ++r) {
            int row = m0 + mt * 16 + 8 * sel + r;
            if (MODE == 2) {
                size_t flat = win_token_to_flat(row) * CCH;
                #pragma unroll
                for (int nt = 0; nt < 4; ++nt) {
                    int col = n0 + nt * 16 + lm;
                    float v = acc[mt][nt][r] + bias[col];
                    outf[flat + col] = resid[flat + col] + v;
                }
            } else {
                #pragma unroll
                for (int nt = 0; nt < 4; ++nt) {
                    int col = n0 + nt * 16 + lm;
                    float v = acc[mt][nt][r] + bias[col];
                    if (MODE == 1)
                        v = 0.5f * v * (1.0f + erff(v * 0.70710678118654752f));
                    if (MODE == 3) {
                        outf[(size_t)row * N + col] = resid[(size_t)row * N + col] + v;
                    } else {
                        outb[(size_t)row * N + col] = f2bf(v);
                    }
                }
            }
        }
    }
}

// ---------------------------------------------------------------------
// Fused window attention: one block per (window, head), 7 waves.
// Wave i owns query rows [16i,16i+16): 7 WMMA for QK^T, softmax via
// half-wave shuffles, P staged bf16 in LDS, 8 WMMA for P.V from LDS.
// ---------------------------------------------------------------------
__global__ __launch_bounds__(224) void attn_kernel(
    const unsigned short* __restrict__ qkv,   // (TOK+pad) x 768 bf16
    const float* __restrict__ rpb,            // 507 x 8
    const int*   __restrict__ rel_idx,        // 98 x 98
    const float* __restrict__ mask,           // 256 x 98 x 98
    unsigned short* __restrict__ out)         // TOK x 256 bf16
{
    __shared__ __attribute__((aligned(16))) unsigned short vt[32 * 128];
    __shared__ __attribute__((aligned(16))) unsigned short pbuf[7 * 16 * 128];

    int b_   = blockIdx.x;
    int head = blockIdx.y;
    int tid  = threadIdx.x;
    int wave = tid >> 5;
    int lane = tid & 31;
    int sel  = lane >> 4, lm = lane & 15;

    // Stage V^T (hd x Kpad) so P.V B-fragments read contiguous K runs.
    for (int idx = tid; idx < 32 * 128; idx += 224) {
        int n = idx >> 7, j = idx & 127;
        unsigned short val = 0;
        if (j < NWTOK)
            val = qkv[(size_t)(b_ * NWTOK + j) * 768 + 512 + head * HDIM + n];
        vt[n * 128 + j] = val;
    }
    __syncthreads();

    // S = Q @ K^T : K=hd=32 -> single WMMA per 16x16 tile.
    v8f acc[7] = {};
    const unsigned short* qrow =
        qkv + (size_t)(b_ * NWTOK + wave * 16 + lm) * 768 + head * HDIM + sel * 8;
    Frag af;
    af.q[0] = *(const u32x4*)qrow;
    af.q[1] = *(const u32x4*)(qrow + 16);
    #pragma unroll
    for (int jt = 0; jt < 7; ++jt) {
        const unsigned short* krow =
            qkv + (size_t)(b_ * NWTOK + jt * 16 + lm) * 768 + 256 + head * HDIM + sel * 8;
        Frag bf;
        bf.q[0] = *(const u32x4*)krow;
        bf.q[1] = *(const u32x4*)(krow + 16);
        acc[jt] = __builtin_amdgcn_wmma_f32_16x16x32_bf16(
            false, af.v, false, bf.v, (short)0, acc[jt], false, false);
    }

    // scale + rel-pos bias + shift mask; pad lanes forced to -inf.
    const float scale = 0.17677669529663687f;   // 32^-0.5
    int mwin = b_ & 255;
    #pragma unroll
    for (int jt = 0; jt < 7; ++jt) {
        int j = jt * 16 + lm;
        #pragma unroll
        for (int r = 0; r < 8; ++r) {
            int i = wave * 16 + 8 * sel + r;
            float v;
            if (i < NWTOK && j < NWTOK) {
                v = acc[jt][r] * scale
                  + rpb[rel_idx[i * NWTOK + j] * NHEAD + head]
                  + mask[((size_t)mwin * NWTOK + i) * NWTOK + j];
            } else {
                v = -1e30f;
            }
            acc[jt][r] = v;
        }
    }

    // Row softmax: each row lives in one 16-lane half (C layout),
    // masks 1/2/4/8 keep the two halves (rows r and r+8) independent.
    #pragma unroll
    for (int r = 0; r < 8; ++r) {
        float mx = -1e30f;
        #pragma unroll
        for (int jt = 0; jt < 7; ++jt) mx = fmaxf(mx, acc[jt][r]);
        #pragma unroll
        for (int m = 1; m < 16; m <<= 1) mx = fmaxf(mx, __shfl_xor(mx, m, 32));
        float sm = 0.f;
        #pragma unroll
        for (int jt = 0; jt < 7; ++jt) {
            float e = __expf(acc[jt][r] - mx);
            acc[jt][r] = e; sm += e;
        }
        #pragma unroll
        for (int m = 1; m < 16; m <<= 1) sm += __shfl_xor(sm, m, 32);
        float inv = 1.0f / fmaxf(sm, 1e-20f);
        #pragma unroll
        for (int jt = 0; jt < 7; ++jt) acc[jt][r] *= inv;
    }

    // P -> LDS as bf16 (rows padded to K=128 with zeros).
    unsigned short* prow = pbuf + (size_t)wave * 16 * 128;
    #pragma unroll
    for (int r = 0; r < 8; ++r) {
        int il = 8 * sel + r;
        #pragma unroll
        for (int jt = 0; jt < 7; ++jt)
            prow[il * 128 + jt * 16 + lm] = f2bf(acc[jt][r]);
        prow[il * 128 + 112 + lm] = 0;
    }
    // same-wave LDS RAW: compiler inserts s_wait_dscnt.

    // O = P @ V : 2 N-tiles x 4 K-steps of 32.
    #pragma unroll
    for (int nt = 0; nt < 2; ++nt) {
        v8f acc2 = {};
        #pragma unroll
        for (int kt = 0; kt < 4; ++kt) {
            int kk = kt * 32;
            Frag pf, vf;
            const unsigned short* pr = prow + lm * 128 + kk + sel * 8;
            pf.q[0] = *(const u32x4*)pr;
            pf.q[1] = *(const u32x4*)(pr + 16);
            const unsigned short* vr = vt + (nt * 16 + lm) * 128 + kk + sel * 8;
            vf.q[0] = *(const u32x4*)vr;
            vf.q[1] = *(const u32x4*)(vr + 16);
            acc2 = __builtin_amdgcn_wmma_f32_16x16x32_bf16(
                false, pf.v, false, vf.v, (short)0, acc2, false, false);
        }
        #pragma unroll
        for (int r = 0; r < 8; ++r) {
            int i = wave * 16 + 8 * sel + r;
            if (i < NWTOK) {
                int col = head * HDIM + nt * 16 + lm;
                out[(size_t)(b_ * NWTOK + i) * CCH + col] = f2bf(acc2[r]);
            }
        }
    }
}

// ---------------------------------------------------------------------
// Host-side orchestration (graph-capture safe: launches only).
// Workspace layout (reuse; total 256 MiB):
//   [0, 1.5MB)    bf16 weights
//   [2MB, 28MB)   bufA : win -> attn_out
//   [28MB,106MB)  bufB : qkv -> h2(LN2)
//   [106MB,158MB) x2   : fp32 residual stream after attention
//   [158MB,256MB) bufH : MLP hidden (bf16)
// ---------------------------------------------------------------------
extern "C" void kernel_launch(void* const* d_in, const int* in_sizes, int n_in,
                              void* d_out, int out_size, void* d_ws, size_t ws_size,
                              hipStream_t stream) {
    (void)in_sizes; (void)n_in; (void)out_size; (void)ws_size;
    const float* x      = (const float*)d_in[0];
    const float* qkv_w  = (const float*)d_in[1];
    const float* qkv_b  = (const float*)d_in[2];
    const float* proj_w = (const float*)d_in[3];
    const float* proj_b = (const float*)d_in[4];
    const float* rpb    = (const float*)d_in[5];
    const float* ln1_g  = (const float*)d_in[6];
    const float* ln1_b  = (const float*)d_in[7];
    const float* ln2_g  = (const float*)d_in[8];
    const float* ln2_b  = (const float*)d_in[9];
    const float* fc1_w  = (const float*)d_in[10];
    const float* fc1_b  = (const float*)d_in[11];
    const float* fc2_w  = (const float*)d_in[12];
    const float* fc2_b  = (const float*)d_in[13];
    const int*   relidx = (const int*)d_in[14];
    const float* amask  = (const float*)d_in[15];
    float* outp = (float*)d_out;

    char* ws = (char*)d_ws;
    unsigned short* w_qkv  = (unsigned short*)(ws + 0);
    unsigned short* w_proj = (unsigned short*)(ws + 393216);
    unsigned short* w_fc1  = (unsigned short*)(ws + 524288);
    unsigned short* w_fc2  = (unsigned short*)(ws + 1048576);
    unsigned short* bufA   = (unsigned short*)(ws + (2ull   << 20));
    unsigned short* bufB   = (unsigned short*)(ws + (28ull  << 20));
    float*          x2     = (float*)         (ws + (106ull << 20));
    unsigned short* bufH   = (unsigned short*)(ws + (158ull << 20));

    // 1) weights -> bf16
    f2bf_kernel<<<(768 * 256  + 255) / 256, 256, 0, stream>>>(qkv_w,  w_qkv,  768 * 256);
    f2bf_kernel<<<(256 * 256  + 255) / 256, 256, 0, stream>>>(proj_w, w_proj, 256 * 256);
    f2bf_kernel<<<(1024 * 256 + 255) / 256, 256, 0, stream>>>(fc1_w,  w_fc1,  1024 * 256);
    f2bf_kernel<<<(256 * 1024 + 255) / 256, 256, 0, stream>>>(fc2_w,  w_fc2,  256 * 1024);

    // 2) LN1 + shift + window partition -> bufA (bf16)
    ln_kernel<true><<<TOK / 8, 256, 0, stream>>>(x, ln1_g, ln1_b, bufA);

    // 3) QKV GEMM: 50176 x 768 x 256 -> bufB
    {   dim3 g(TOK / 128, 768 / 64);
        gemm_bf16<0><<<g, 128, 0, stream>>>(bufA, w_qkv, qkv_b,
                                            TOK, 768, 256, bufB, nullptr, nullptr); }

    // 4) fused window attention -> bufA (reused)
    {   dim3 g(512, NHEAD);
        attn_kernel<<<g, 224, 0, stream>>>(bufB, rpb, relidx, amask, bufA); }

    // 5) proj GEMM + window reverse + unshift + residual -> x2 (fp32)
    {   dim3 g(TOK / 128, 256 / 64);
        gemm_bf16<2><<<g, 128, 0, stream>>>(bufA, w_proj, proj_b,
                                            TOK, 256, 256, nullptr, x2, x); }

    // 6) LN2 -> bufB (reused, bf16)
    ln_kernel<false><<<TOK / 8, 256, 0, stream>>>(x2, ln2_g, ln2_b, bufB);

    // 7) fc1 + exact GELU -> bufH
    {   dim3 g(TOK / 128, 1024 / 64);
        gemm_bf16<1><<<g, 128, 0, stream>>>(bufB, w_fc1, fc1_b,
                                            TOK, 1024, 256, bufH, nullptr, nullptr); }

    // 8) fc2 + residual -> d_out (fp32)
    {   dim3 g(TOK / 128, 256 / 64);
        gemm_bf16<3><<<g, 128, 0, stream>>>(bufH, w_fc2, fc2_b,
                                            TOK, 256, 1024, nullptr, outp, x2); }
}
